// ScaledDotProductAttention_89386859364688
// MI455X (gfx1250) — compile-verified, matrix operand order checked by reading, not run on hardware
//
#include <hip/hip_runtime.h>
#include <hip/hip_bf16.h>

// ---------------------------------------------------------------------------
// Fused scaled-dot-product attention for MI455X (gfx1250, wave32, WMMA).
//   scores = q @ v^T / sqrt(128); attn = softmax(scores); out = attn^T @ v
// Outputs (concatenated in d_out): out [8,2048,128] f32, attn [8,2048,2048] f32.
//
// bf16x3 error-compensated WMMA (v_wmma_f32_16x16x32_bf16) for both GEMMs,
// with 3 independent accumulator chains for WMMA ILP.  q/v pre-split ONCE
// into bf16 hi/lo planes (+ transposed v) so hot loops are b128 loads + WMMA.
// Pass 2 double-buffers the shared q chunk in LDS via
// global_load_async_to_lds_b128 (ASYNCcnt) overlapped with compute, and
// streams attn/out with non-temporal stores to protect L2 residency.
// ---------------------------------------------------------------------------

typedef __attribute__((ext_vector_type(16))) __bf16 v16bf;
typedef __attribute__((ext_vector_type(8)))  __bf16 v8bf;
typedef __attribute__((ext_vector_type(8)))  float  v8f;

constexpr int   B_  = 8;
constexpr int   L_  = 2048;
constexpr int   D_  = 128;
constexpr float INV_TEMP = 0.08838834764831845f;  // 1/sqrt(128)
constexpr int   QPAD = 8;                          // LDS row pad (elements)

static __device__ __forceinline__ v8f wmma_bf16(v16bf a, v16bf b, v8f c) {
    return __builtin_amdgcn_wmma_f32_16x16x32_bf16(
        false, a, false, b, (short)0, c, false, false);
}

// Load one bf16 WMMA operand (v16bf) from a plane pointer positioned at
// (row_base + chunk*32 + g*8): halves 0..7 at +0, halves 8..15 at +16 elems.
static __device__ __forceinline__ v16bf load16(const __bf16* p) {
    union { v16bf v; v8bf h[2]; } u;
    u.h[0] = *(const v8bf*)(p);
    u.h[1] = *(const v8bf*)(p + 16);
    return u.v;
}

static __device__ __forceinline__ void split_bf16(float x, __bf16& hi, __bf16& lo) {
    hi = (__bf16)x;
    lo = (__bf16)(x - (float)hi);
}

// ---------------------------------------------------------------------------
// Pre-pass A: elementwise split of q into bf16 hi/lo planes.
// ---------------------------------------------------------------------------
__global__ void __launch_bounds__(256)
split_q_kernel(const float* __restrict__ src,
               __bf16* __restrict__ dhi, __bf16* __restrict__ dlo, int n)
{
    int i = blockIdx.x * 256 + threadIdx.x;
    if (i < n) {
        __bf16 hi, lo; split_bf16(src[i], hi, lo);
        dhi[i] = hi; dlo[i] = lo;
    }
}

// ---------------------------------------------------------------------------
// Pre-pass B: split v into hi/lo planes AND transposed planes vT[b, d, l]
// via a 32x32 LDS tile transpose (coalesced both ways).
// Grid: B * (L/32) * (D/32) blocks of 256 threads.
// ---------------------------------------------------------------------------
__global__ void __launch_bounds__(256)
split_transpose_v_kernel(const float* __restrict__ v,
                         __bf16* __restrict__ vhi, __bf16* __restrict__ vlo,
                         __bf16* __restrict__ vthi, __bf16* __restrict__ vtlo)
{
    __shared__ __bf16 thi[32][33];
    __shared__ __bf16 tlo[32][33];

    const int tiles_l = L_ / 32, tiles_d = D_ / 32;
    const int b  = blockIdx.x / (tiles_l * tiles_d);
    const int r0 = blockIdx.x % (tiles_l * tiles_d);
    const int l0 = (r0 / tiles_d) * 32;
    const int d0 = (r0 % tiles_d) * 32;
    const int col = threadIdx.x & 31;
    const int ry  = threadIdx.x >> 5;      // 0..7

#pragma unroll
    for (int k = 0; k < 4; ++k) {
        int row = ry * 4 + k;              // 0..31
        size_t idx = ((size_t)b * L_ + l0 + row) * D_ + d0 + col;
        __bf16 hi, lo; split_bf16(v[idx], hi, lo);
        vhi[idx] = hi; vlo[idx] = lo;
        thi[col][row] = hi;                // transposed staging
        tlo[col][row] = lo;
    }
    __syncthreads();
#pragma unroll
    for (int k = 0; k < 4; ++k) {
        int r = ry * 4 + k;                // d offset within tile
        size_t idx = ((size_t)b * D_ + d0 + r) * L_ + l0 + col;
        vthi[idx] = thi[r][col];
        vtlo[idx] = tlo[r][col];
    }
}

// ---------------------------------------------------------------------------
// Pass 1: per-row softmax stats (row max m_i, reciprocal row sum 1/l_i).
// Grid: B*(L/16) blocks of 128 threads; each wave: one 16-row i-tile, one
// quarter of the j range, online (max,sum); waves combined via LDS.
// ---------------------------------------------------------------------------
__global__ void __launch_bounds__(128)
attn_stats_kernel(const __bf16* __restrict__ qhi, const __bf16* __restrict__ qlo,
                  const __bf16* __restrict__ vhi, const __bf16* __restrict__ vlo,
                  float* __restrict__ Mrow, float* __restrict__ Linv)
{
    const int b    = blockIdx.x / (L_ / 16);
    const int i0   = (blockIdx.x % (L_ / 16)) * 16;
    const int lane = threadIdx.x & 31;
    const int wave = threadIdx.x >> 5;
    const int g    = lane >> 4;
    const int ml   = lane & 15;

    const size_t qoff = ((size_t)b * L_ + i0 + ml) * D_ + g * 8;
    v16bf ahi[4], alo[4];
#pragma unroll
    for (int c = 0; c < 4; ++c) {
        ahi[c] = load16(qhi + qoff + c * 32);
        alo[c] = load16(qlo + qoff + c * 32);
    }

    float mrun[8], lrun[8];
#pragma unroll
    for (int r = 0; r < 8; ++r) { mrun[r] = -3.0e38f; lrun[r] = 0.0f; }

    const v8f Z = {0.f, 0.f, 0.f, 0.f, 0.f, 0.f, 0.f, 0.f};
    const int jbeg = wave * (L_ / 4);
#pragma unroll 1
    for (int j0 = jbeg; j0 < jbeg + (L_ / 4); j0 += 16) {
        const size_t voff = ((size_t)b * L_ + j0 + ml) * D_ + g * 8;
        // 3 independent accumulator chains (depth 4 each) for WMMA ILP.
        v8f S0 = Z, S1 = Z, S2 = Z;
#pragma unroll
        for (int c = 0; c < 4; ++c) {
            v16bf bh = load16(vhi + voff + c * 32);
            v16bf bl = load16(vlo + voff + c * 32);
            S0 = wmma_bf16(ahi[c], bh, S0);
            S1 = wmma_bf16(ahi[c], bl, S1);
            S2 = wmma_bf16(alo[c], bh, S2);
        }
        v8f S = (S0 + S1) + S2;
#pragma unroll
        for (int r = 0; r < 8; ++r) {
            float s = S[r] * INV_TEMP;
            float rmax = s;
#pragma unroll
            for (int m = 1; m < 16; m <<= 1)
                rmax = fmaxf(rmax, __shfl_xor(rmax, m, 32));
            float mnew = fmaxf(mrun[r], rmax);
            float p = __expf(s - mnew);
#pragma unroll
            for (int m = 1; m < 16; m <<= 1)
                p += __shfl_xor(p, m, 32);
            lrun[r] = lrun[r] * __expf(mrun[r] - mnew) + p;
            mrun[r] = mnew;
        }
    }

    __shared__ float sm[4][16];
    __shared__ float sl[4][16];
    if (ml == 0) {
#pragma unroll
        for (int r = 0; r < 8; ++r) {
            sm[wave][g * 8 + r] = mrun[r];
            sl[wave][g * 8 + r] = lrun[r];
        }
    }
    __syncthreads();
    if (threadIdx.x < 16) {
        int row = threadIdx.x;
        float m = sm[0][row];
#pragma unroll
        for (int w = 1; w < 4; ++w) m = fmaxf(m, sm[w][row]);
        float l = 0.0f;
#pragma unroll
        for (int w = 0; w < 4; ++w) l += sl[w][row] * __expf(sm[w][row] - m);
        Mrow[(size_t)b * L_ + i0 + row] = m;
        Linv[(size_t)b * L_ + i0 + row] = 1.0f / l;
    }
}

// ---------------------------------------------------------------------------
// Pass 2: fused attn-write + out accumulation.  Grid: B*(L/64) blocks of 128.
// Each wave owns a 16-column j-block (matmul-1 B register-resident).  The q
// chunk (shared by all 4 waves) is DOUBLE-BUFFERED in LDS: async copy of
// chunk n+1 (global_load_async_to_lds_b128) overlaps WMMA compute of chunk n.
// Score-tile C layout feeds matmul-2's bf16 A operand with zero cross-lane
// data movement.  attn/out leave via non-temporal stores.
// ---------------------------------------------------------------------------
__global__ void __launch_bounds__(128)
attn_out_kernel(const __bf16* __restrict__ qhi, const __bf16* __restrict__ qlo,
                const __bf16* __restrict__ vhi, const __bf16* __restrict__ vlo,
                const __bf16* __restrict__ vthi, const __bf16* __restrict__ vtlo,
                const float* __restrict__ Mrow, const float* __restrict__ Linv,
                float* __restrict__ outp, float* __restrict__ attnp)
{
    __shared__ alignas(16) __bf16 qs_hi[2][32][D_ + QPAD];
    __shared__ alignas(16) __bf16 qs_lo[2][32][D_ + QPAD];

    const int b    = blockIdx.x / (L_ / 64);
    const int jt   = blockIdx.x % (L_ / 64);
    const int lane = threadIdx.x & 31;
    const int wave = threadIdx.x >> 5;
    const int j0   = jt * 64 + wave * 16;
    const int g    = lane >> 4;
    const int ml   = lane & 15;
    const int tid  = threadIdx.x;

    // Issue the cooperative async copy of one 32-row q chunk into buffer buf.
    auto stage = [&](int i0s, int buf) {
        unsigned lhi = (unsigned)(uintptr_t)&qs_hi[buf][0][0];
        unsigned llo = (unsigned)(uintptr_t)&qs_lo[buf][0][0];
#pragma unroll
        for (int u = 0; u < 4; ++u) {
            int unit = tid + u * 128;      // 512 b128 units per plane
            int row  = unit >> 4;
            int cb   = unit & 15;
            unsigned d = (unsigned)(row * (D_ + QPAD) + cb * 8) * 2u;
            const __bf16* ghi = qhi + ((size_t)b * L_ + i0s + row) * D_ + cb * 8;
            const __bf16* glo = qlo + ((size_t)b * L_ + i0s + row) * D_ + cb * 8;
            asm volatile("global_load_async_to_lds_b128 %0, %1, off"
                         :: "v"(lhi + d), "v"(ghi) : "memory");
            asm volatile("global_load_async_to_lds_b128 %0, %1, off"
                         :: "v"(llo + d), "v"(glo) : "memory");
        }
    };

    // Matmul-1 B operand: v rows j0..j0+15, register-resident for the i sweep.
    const size_t voff = ((size_t)b * L_ + j0 + ml) * D_ + g * 8;
    v16bf bhi[4], blo[4];
#pragma unroll
    for (int c = 0; c < 4; ++c) {
        bhi[c] = load16(vhi + voff + c * 32);
        blo[c] = load16(vlo + voff + c * 32);
    }

    const v8f Z = {0.f, 0.f, 0.f, 0.f, 0.f, 0.f, 0.f, 0.f};
    v8f acc[8];
#pragma unroll
    for (int db = 0; db < 8; ++db) acc[db] = Z;

    stage(0, 0);                           // prologue: fill buffer 0

#pragma unroll 1
    for (int i0 = 0; i0 < L_; i0 += 32) {
        const int buf = (i0 >> 5) & 1;
        // Current buffer's copy complete in all waves; also proves everyone
        // finished reading the other buffer last iteration.
        asm volatile("s_wait_asynccnt 0" ::: "memory");
        __syncthreads();
        if (i0 + 32 < L_) stage(i0 + 32, buf ^ 1);   // overlap next copy

        // ---- matmul 1: two 16x16 score tiles (A via ds_load_b128).
        v8f S01[2];
#pragma unroll
        for (int t = 0; t < 2; ++t) {
            v8f S0 = Z, S1 = Z, S2 = Z;    // 3 independent WMMA chains
#pragma unroll
            for (int c = 0; c < 4; ++c) {
                v16bf ah = load16(&qs_hi[buf][t * 16 + ml][c * 32 + g * 8]);
                v16bf al = load16(&qs_lo[buf][t * 16 + ml][c * 32 + g * 8]);
                S0 = wmma_bf16(ah, bhi[c], S0);
                S1 = wmma_bf16(ah, blo[c], S1);
                S2 = wmma_bf16(al, bhi[c], S2);
            }
            S01[t] = (S0 + S1) + S2;
        }

        // ---- softmax apply + attn store (NT) + pack matmul-2 A operand.
        v16bf phi, plo;
#pragma unroll
        for (int t = 0; t < 2; ++t) {
            const size_t sbase = (size_t)b * L_ + i0 + t * 16 + g * 8;
            float4 m0 = *(const float4*)(Mrow + sbase);
            float4 m1 = *(const float4*)(Mrow + sbase + 4);
            float4 l0 = *(const float4*)(Linv + sbase);
            float4 l1 = *(const float4*)(Linv + sbase + 4);
            float mr[8] = {m0.x, m0.y, m0.z, m0.w, m1.x, m1.y, m1.z, m1.w};
            float lr[8] = {l0.x, l0.y, l0.z, l0.w, l1.x, l1.y, l1.z, l1.w};
#pragma unroll
            for (int r = 0; r < 8; ++r) {
                int i = i0 + t * 16 + r + g * 8;
                float p = __expf(S01[t][r] * INV_TEMP - mr[r]) * lr[r];
                __builtin_nontemporal_store(
                    p, &attnp[((size_t)b * L_ + i) * L_ + j0 + ml]);
                __bf16 hi, lo; split_bf16(p, hi, lo);
                phi[t * 8 + r] = hi; plo[t * 8 + r] = lo;
            }
        }

        // ---- matmul 2: out[j,d] += attn^T * v, K=32; 8 independent chains.
#pragma unroll
        for (int db = 0; db < 8; ++db) {
            const size_t toff = ((size_t)b * D_ + db * 16 + ml) * L_ + i0 + g * 8;
            v16bf vh = load16(vthi + toff);
            v16bf vl = load16(vtlo + toff);
            acc[db] = wmma_bf16(phi, vh, acc[db]);
            acc[db] = wmma_bf16(phi, vl, acc[db]);
            acc[db] = wmma_bf16(plo, vh, acc[db]);
        }
    }

    // Store out tile (NT): C layout lane = d col (contiguous), VGPR = j row.
#pragma unroll
    for (int db = 0; db < 8; ++db)
#pragma unroll
        for (int r = 0; r < 8; ++r)
            __builtin_nontemporal_store(
                acc[db][r],
                &outp[((size_t)b * L_ + j0 + r + g * 8) * D_ + db * 16 + ml]);
}

// ---------------------------------------------------------------------------
extern "C" void kernel_launch(void* const* d_in, const int* in_sizes, int n_in,
                              void* d_out, int out_size, void* d_ws, size_t ws_size,
                              hipStream_t stream) {
    const float* q = (const float*)d_in[0];
    // d_in[1] (k) unused — faithful to the reference, which ignores k.
    const float* v = (const float*)d_in[2];

    float* outp  = (float*)d_out;                          // [B, L, D]
    float* attnp = (float*)d_out + (size_t)B_ * L_ * D_;   // [B, L, L]

    const size_t NE = (size_t)B_ * L_ * D_;   // elements per tensor (2M)
    char* w = (char*)d_ws;
    float*  Mrow = (float*)w;   w += (size_t)B_ * L_ * sizeof(float);
    float*  Linv = (float*)w;   w += (size_t)B_ * L_ * sizeof(float);
    __bf16* qhi  = (__bf16*)w;  w += NE * 2;
    __bf16* qlo  = (__bf16*)w;  w += NE * 2;
    __bf16* vhi  = (__bf16*)w;  w += NE * 2;
    __bf16* vlo  = (__bf16*)w;  w += NE * 2;
    __bf16* vthi = (__bf16*)w;  w += NE * 2;
    __bf16* vtlo = (__bf16*)w;  // total ws use: ~24.1 MB

    split_q_kernel<<<dim3((unsigned)((NE + 255) / 256)), dim3(256), 0, stream>>>(
        q, qhi, qlo, (int)NE);
    split_transpose_v_kernel<<<dim3(B_ * (L_ / 32) * (D_ / 32)), dim3(256), 0, stream>>>(
        v, vhi, vlo, vthi, vtlo);
    attn_stats_kernel<<<dim3(B_ * (L_ / 16)), dim3(128), 0, stream>>>(
        qhi, qlo, vhi, vlo, Mrow, Linv);
    attn_out_kernel<<<dim3(B_ * (L_ / 64)), dim3(128), 0, stream>>>(
        qhi, qlo, vhi, vlo, vthi, vtlo, Mrow, Linv, outp, attnp);
}
